// MultiHeadNeighborhoodAttentionModule_6923487282084
// MI455X (gfx1250) — matrix operand order repaired and checked by reading, not compiled, not run on hardware
//
#include <hip/hip_runtime.h>
#include <hip/hip_bf16.h>

#define EMB   256
#define ADIM  64
#define NH    2
#define NK    16
#define TN    512            // NH * EMB
#define TILE  16
#define CR    2              // batch rows per async chunk
#define NCH   (TILE / CR)    // 8 chunks per tile
#define NB_STR 260           // padded f32 stride per neighbor row (260 % 64 == 4 banks)
#define CB_STR (EMB + 8)     // padded LDS stride (ushort) for center bf16
#define CX_STR (TN + 8)      // padded LDS stride (ushort) for ctx bf16

typedef __attribute__((ext_vector_type(16))) __bf16 v16bf;
typedef __attribute__((ext_vector_type(8)))  float  v8f;

union FragBF { v16bf bf; unsigned int u[8]; };

__device__ __forceinline__ unsigned short f32_to_bf16_rne(float f) {
    unsigned int u = __builtin_bit_cast(unsigned int, f);
    u += 0x7FFFu + ((u >> 16) & 1u);
    return (unsigned short)(u >> 16);
}

// Async DMA one chunk (CR rows x 16 neighbors x 256 f32 = 32KB) into LDS.
// 16 x global_load_async_to_lds_b128 per wave, per-lane addresses, ASYNCcnt-tracked.
__device__ __forceinline__ void issue_chunk_async(const float* __restrict__ neigh,
                                                  int b0, int c, float* s_buf, int tid) {
#pragma unroll
    for (int it = 0; it < 16; ++it) {
        int i   = tid + it * 128;          // vec4 index within chunk: 0..2047
        int r2  = i >> 10;                 // row-in-chunk (1024 vec4 per row)
        int rem = i & 1023;
        int k   = rem >> 6;                // neighbor index
        int d4  = rem & 63;                // vec4 index within 256-dim row
        unsigned lds_off = (unsigned)(size_t)&s_buf[(r2 * NK + k) * NB_STR + d4 * 4];
        const float* g = neigh + ((size_t)((b0 + c * CR + r2) * NK + k) * EMB + d4 * 4);
        asm volatile("global_load_async_to_lds_b128 %0, %1, off"
                     :: "v"(lds_off), "v"((unsigned long long)(size_t)g)
                     : "memory");
    }
}

#define WAIT_ASYNC_16() asm volatile("s_wait_asynccnt 16" ::: "memory")
#define WAIT_ASYNC_0()  asm volatile("s_wait_asynccnt 0"  ::: "memory")

// ---------------------------------------------------------------------------
// Prep: M^T[n][d] = sum_a Wq[h][a][d] * Wk[h][a][j]  (n = h*256 + j), bf16.
// Also convert W_out ([256 out][512 in] = B^T layout) to bf16. Both live in
// d_ws (512KB) and stay resident in the 192MB L2 for the whole main kernel.
// ---------------------------------------------------------------------------
__global__ void prep_kernel(const float* __restrict__ Wq,
                            const float* __restrict__ Wk,
                            const float* __restrict__ Wout,
                            unsigned short* __restrict__ McatT,
                            unsigned short* __restrict__ WoutBf) {
    int idx = blockIdx.x * blockDim.x + threadIdx.x;   // 0 .. 131071
    if (idx >= TN * EMB) return;
    int n = idx >> 8;
    int d = idx & 255;
    int h = n >> 8;
    int j = n & 255;
    const float* wq = Wq + (size_t)h * ADIM * EMB;
    const float* wk = Wk + (size_t)h * ADIM * EMB;
    float acc = 0.f;
#pragma unroll 8
    for (int a = 0; a < ADIM; ++a)
        acc = fmaf(wq[a * EMB + d], wk[a * EMB + j], acc);
    McatT[idx]  = f32_to_bf16_rne(acc);
    WoutBf[idx] = f32_to_bf16_rne(Wout[idx]);
}

// ---------------------------------------------------------------------------
// Main kernel: one 16-row batch tile per 128-thread block (4 waves).
// ---------------------------------------------------------------------------
__global__ __launch_bounds__(128)
void attn_kernel(const float* __restrict__ center,
                 const float* __restrict__ neigh,
                 const float* __restrict__ nw,
                 const unsigned char* __restrict__ valid,
                 const float* __restrict__ b_out,
                 const unsigned short* __restrict__ McatT,
                 const unsigned short* __restrict__ WoutBf,
                 float* __restrict__ out) {
    __shared__ unsigned short s_cbf[TILE * CB_STR];        // 8.3KB  center bf16
    __shared__ float          s_T[TILE * TN];              // 32KB   T = c @ [M0|M1]
    __shared__ unsigned short s_ctx[TILE * CX_STR];        // 16.3KB multi_ctx bf16
    __shared__ float          s_attn[CR * NH * NK];        // per-chunk attention
    __shared__ float          s_nb[2][CR * NK * NB_STR];   // 2 x 32.5KB neighbor chunks

    const int tid  = threadIdx.x;
    const int wave = tid >> 5;
    const int lane = tid & 31;
    const int half = lane >> 4;
    const int l16  = lane & 15;
    const int b0   = blockIdx.x * TILE;

    // Kick off DMA of neighbor chunk 0 immediately (overlaps with GEMM1).
    issue_chunk_async(neigh, b0, 0, s_nb[0], tid);

    // ---- Phase 1: stage center tile as bf16 (coalesced f32 read) ----------
    for (int i = tid; i < TILE * EMB; i += 128) {
        int m = i >> 8, d = i & 255;
        s_cbf[m * CB_STR + d] = f32_to_bf16_rne(center[(size_t)b0 * EMB + i]);
    }
    __syncthreads();

    // ---- Phase 2: T[16][512] = center_bf16 @ M  (B fragments from L2) -----
    for (int nt = wave; nt < TN / 16; nt += 4) {
        v8f acc = {};
        for (int ks = 0; ks < EMB / 32; ++ks) {
            FragBF a, b;
#pragma unroll
            for (int v = 0; v < 8; ++v) {
                int ka = ks * 32 + ((v & 4) << 2) + half * 8 + 2 * (v & 3);
                a.u[v] = *(const unsigned int*)&s_cbf[l16 * CB_STR + ka];
                int kb = ks * 32 + ((v & 4) << 2) + half * 8 + 2 * (v & 3);
                b.u[v] = *(const unsigned int*)&McatT[(size_t)(nt * 16 + l16) * EMB + kb];
            }
            acc = __builtin_amdgcn_wmma_f32_16x16x32_bf16(false, a.bf, false, b.bf,
                                                          (short)0, acc, false, false);
        }
#pragma unroll
        for (int v = 0; v < 8; ++v)
            s_T[(v + half * 8) * TN + nt * 16 + l16] = acc[v];
    }
    __syncthreads();

    // ---- Phase 3: chunked, double-buffered scores/softmax/ctx -------------
    for (int c = 0; c < NCH; ++c) {
        if (c + 1 < NCH) {
            issue_chunk_async(neigh, b0, c + 1, s_nb[(c + 1) & 1], tid);
            WAIT_ASYNC_16();   // in-order retirement => chunk c fully in LDS
        } else {
            WAIT_ASYNC_0();
        }
        __syncthreads();
        const float* nb = s_nb[c & 1];

        // scores + softmax: 64 dots (2 rows x 2 heads x 16 k), lanes 0..15 of
        // each 16-lane group hold all k of one (row,head) -> shfl reductions.
        if (tid < CR * NH * NK) {
            int r2 = tid >> 5, h = (tid >> 4) & 1, k = tid & 15;
            int r  = c * CR + r2;
            const float* np = nb + (r2 * NK + k) * NB_STR;
            const float* tp = s_T + r * TN + h * EMB;
            float s = 0.f;
#pragma unroll 8
            for (int j = 0; j < EMB; ++j)
                s = fmaf(np[j], tp[j], s);
            s *= 0.125f;                                   // 1/sqrt(64)
            float m = s;
#pragma unroll
            for (int o = 8; o > 0; o >>= 1) m = fmaxf(m, __shfl_xor(m, o, 16));
            float e = __expf(s - m);
            float sum = e;
#pragma unroll
            for (int o = 8; o > 0; o >>= 1) sum += __shfl_xor(sum, o, 16);
            float a1 = (e / sum) * nw[(size_t)(b0 + r) * NK + k];
            float s2 = a1;
#pragma unroll
            for (int o = 8; o > 0; o >>= 1) s2 += __shfl_xor(s2, o, 16);
            s_attn[(r2 * NH + h) * NK + k] = a1 / (s2 + 1e-8f);
        }
        __syncthreads();

        // ctx: 2 rows x 512 cols, 16 LDS FMAs each, write bf16 multi_ctx.
        for (int o = tid; o < CR * TN; o += 128) {
            int r2 = o >> 9, cc = o & 511, h = cc >> 8, d = cc & 255;
            const float* np = nb + r2 * NK * NB_STR + d;
            const float* ap = s_attn + (r2 * NH + h) * NK;
            float acc = 0.f;
#pragma unroll
            for (int k = 0; k < NK; ++k)
                acc = fmaf(ap[k], np[k * NB_STR], acc);
            s_ctx[(c * CR + r2) * CX_STR + cc] = f32_to_bf16_rne(acc);
        }
        __syncthreads();   // all waves done with buffer (c&1) before reuse
    }

    // ---- Phase 4+5: out = ctx @ W_out^T + b_out, masked select, store -----
    for (int nt = wave; nt < EMB / 16; nt += 4) {
        v8f acc = {};
        for (int ks = 0; ks < TN / 32; ++ks) {
            FragBF a, b;
#pragma unroll
            for (int v = 0; v < 8; ++v) {
                int ka = ks * 32 + ((v & 4) << 2) + half * 8 + 2 * (v & 3);
                a.u[v] = *(const unsigned int*)&s_ctx[l16 * CX_STR + ka];
                int kb = ks * 32 + ((v & 4) << 2) + half * 8 + 2 * (v & 3);
                b.u[v] = *(const unsigned int*)&WoutBf[(size_t)(nt * 16 + l16) * TN + kb];
            }
            acc = __builtin_amdgcn_wmma_f32_16x16x32_bf16(false, a.bf, false, b.bf,
                                                          (short)0, acc, false, false);
        }
        int n = nt * 16 + l16;
        float bias = b_out[n];
#pragma unroll
        for (int v = 0; v < 8; ++v) {
            int row = b0 + v + half * 8;
            float val = acc[v] + bias;
            if (!valid[row]) val = center[(size_t)row * EMB + n];
            out[(size_t)row * EMB + n] = val;
        }
    }
}

// ---------------------------------------------------------------------------
extern "C" void kernel_launch(void* const* d_in, const int* in_sizes, int n_in,
                              void* d_out, int out_size, void* d_ws, size_t ws_size,
                              hipStream_t stream) {
    (void)n_in; (void)out_size; (void)ws_size;
    const float* center        = (const float*)d_in[0];
    const float* neigh         = (const float*)d_in[1];
    const float* nw            = (const float*)d_in[2];
    const unsigned char* valid = (const unsigned char*)d_in[3];  // jax bool = 1 byte
    const float* Wq            = (const float*)d_in[4];
    const float* Wk            = (const float*)d_in[5];
    const float* Wout          = (const float*)d_in[6];
    const float* bout          = (const float*)d_in[7];
    float* out = (float*)d_out;

    unsigned short* McatT  = (unsigned short*)d_ws;        // 512*256 bf16 = 256KB
    unsigned short* WoutBf = McatT + (size_t)TN * EMB;     // 256*512 bf16 = 256KB

    int B = in_sizes[0] / EMB;   // 32768

    prep_kernel<<<(TN * EMB + 255) / 256, 256, 0, stream>>>(Wq, Wk, Wout, McatT, WoutBf);
    attn_kernel<<<B / TILE, 128, 0, stream>>>(center, neigh, nw, valid, bout,
                                              McatT, WoutBf, out);
}
